// LinearChainCRF_60722247631331
// MI455X (gfx1250) — matrix operand order repaired and checked by reading, not compiled, not run on hardware
//
#include <hip/hip_runtime.h>
#include <stdint.h>

typedef __attribute__((ext_vector_type(16))) _Float16 v16h;
typedef __attribute__((ext_vector_type(8)))  float    v8f;

#define Bsz 512
#define Tsz 1024
#define Ksz 48

__device__ __forceinline__ int bytesum(unsigned x) {
  return (int)(((x & 0x01010101u) * 0x01010101u) >> 24);
}

__device__ __forceinline__ v8f wmma_f16(v16h a, v16h b, v8f c) {
  return __builtin_amdgcn_wmma_f32_16x16x32_f16(false, a, false, b, (short)0, c,
                                                false, false);
}

// Prefetch one step's emissions (24 f32 as 6 x b128) and target bytes
// (24 u8 as 3 x b64) for this lane's (sequence, j-halfset).
__device__ __forceinline__ void load_step(const float* __restrict__ emL,
                                          const uint8_t* __restrict__ tgL,
                                          int t, float ev[24], uint2 gv[3]) {
#pragma unroll
  for (int g = 0; g < 3; ++g) {
    const float* p = emL + (size_t)t * Ksz + 16 * g;
    const float4 a = *(const float4*)p;
    const float4 b = *(const float4*)(p + 4);
    ev[g * 8 + 0] = a.x; ev[g * 8 + 1] = a.y;
    ev[g * 8 + 2] = a.z; ev[g * 8 + 3] = a.w;
    ev[g * 8 + 4] = b.x; ev[g * 8 + 5] = b.y;
    ev[g * 8 + 6] = b.z; ev[g * 8 + 7] = b.w;
    gv[g] = *(const uint2*)(tgL + (size_t)t * Ksz + 16 * g);
  }
}

// One CRF step: q^T = E^T x p^T (6 WMMA), * exp(em)*gate, capture at len-1,
// power-of-2 renorm, register-space transpose (shfl_xor 16) into next B operand.
__device__ __forceinline__ void do_step(int t, const float ev[24],
                                        const uint2 gv[3],
                                        const v16h (&A)[3][2], v16h& Bc0,
                                        v16h& Bc1, const float ee[24], bool sup,
                                        bool h0, int len, int& ex_acc,
                                        float& res) {
  // emission multipliers (loads already done; gate is a value-select, no branch)
  float ml[24];
#pragma unroll
  for (int g = 0; g < 3; ++g) {
    const unsigned lo = gv[g].x, hi = gv[g].y;
#pragma unroll
    for (int r = 0; r < 8; ++r) {
      const unsigned w = (r < 4) ? lo : hi;
      const float tg = (float)((w >> (8 * (r & 3))) & 0xffu);
      const float gate = sup ? tg : 1.0f;
      ml[g * 8 + r] = __expf(ev[g * 8 + r]) * gate;
    }
  }
  // q'[s][j] for this lane's 24 j-values
  float q[24];
#pragma unroll
  for (int g = 0; g < 3; ++g) {
    v8f acc = {};
    acc = wmma_f16(A[g][0], Bc0, acc);
    acc = wmma_f16(A[g][1], Bc1, acc);
    union { v8f v; float f[8]; } du;
    du.v = acc;
#pragma unroll
    for (int r = 0; r < 8; ++r) q[g * 8 + r] = du.f[r] * ml[g * 8 + r];
  }
  // per-lane reductions + half combine
  float vmax = 1e-30f, dot = 0.f;
#pragma unroll
  for (int i = 0; i < 24; ++i) {
    vmax = fmaxf(vmax, q[i]);
    dot += q[i] * ee[i];
  }
  vmax = fmaxf(vmax, __shfl_xor(vmax, 16, 32));
  dot = dot + __shfl_xor(dot, 16, 32);
  // capture z = logsumexp(alpha_t + end) at t == len-1 (uses pre-update scale)
  const float z = (float)ex_acc * 0.69314718055994531f + __logf(dot);
  res = (t == len - 1) ? z : res;
  // exact power-of-2 renormalization
  int e2;
  (void)frexpf(vmax, &e2);
  ex_acc += e2;
  // pack normalized p into f16 pair dwords: pk[g*4+u] = (j=16g+8h+2u, +1)
  unsigned pk[12];
#pragma unroll
  for (int g = 0; g < 3; ++g)
#pragma unroll
    for (int u = 0; u < 4; ++u) {
      union { unsigned w; _Float16 hh[2]; } pw;
      pw.hh[0] = (_Float16)ldexpf(q[g * 8 + 2 * u], -e2);
      pw.hh[1] = (_Float16)ldexpf(q[g * 8 + 2 * u + 1], -e2);
      pk[g * 4 + u] = pw.w;
    }
  // half-wave exchange: partner lane holds the complementary j-halves
  unsigned sw[12];
#pragma unroll
  for (int i = 0; i < 12; ++i) sw[i] = (unsigned)__shfl_xor((int)pk[i], 16, 32);
  // assemble next B operand (B-layout: lane half selects K+-16 within chunk)
  union { v16h v; unsigned u[8]; } b0, b1;
#pragma unroll
  for (int v = 0; v < 4; ++v) {
    b0.u[v]     = h0 ? pk[0 * 4 + v] : sw[1 * 4 + v];  // k 0..7   / 16..23
    b0.u[v + 4] = h0 ? sw[0 * 4 + v] : pk[1 * 4 + v];  // k 8..15  / 24..31
    b1.u[v]     = h0 ? pk[2 * 4 + v] : 0u;             // k 32..39 / pad
    b1.u[v + 4] = h0 ? sw[2 * 4 + v] : 0u;             // k 40..47 / pad
  }
  Bc0 = b0.v;
  Bc1 = b1.v;
}

__global__ __launch_bounds__(32)
void crf_fwd_wmma(const float* __restrict__ em,       // [B,T,K] f32
                  const uint8_t* __restrict__ maskb,  // [B,T] bool
                  const uint8_t* __restrict__ tgt,    // [B,T,K] bool
                  const float* __restrict__ trans,    // [K,K]
                  const float* __restrict__ startt,   // [K]
                  const float* __restrict__ endt,     // [K]
                  const uint8_t* __restrict__ forb,   // [K,K] bool
                  const uint8_t* __restrict__ sforb,  // [K] bool
                  const uint8_t* __restrict__ eforb,  // [K] bool
                  float* __restrict__ out) {          // [B]
  const int l = threadIdx.x;
  const int s = l & 15;           // row: 0..7 supervised, 8..15 unsupervised
  const int h = l >> 4;           // lane half (j/k half-set selector)
  const bool h0 = (h == 0);
  const bool sup = ((l & 8) == 0);
  const int b_base = blockIdx.x * 8;
  const int bbl = b_base + (l & 7);  // this lane's sequence

  // ---- A = E^T = exp(masked trans)^T, constant, WMMA A-layout ----
  // tile g: out-rows j = 16g + (l&15); chunk c: i = 32c + within(v) + 16h + qq
  v16h A[3][2];
#pragma unroll
  for (int g = 0; g < 3; ++g)
#pragma unroll
    for (int c = 0; c < 2; ++c) {
      union { v16h v; _Float16 hh[16]; } av;
#pragma unroll
      for (int v = 0; v < 8; ++v)
#pragma unroll
        for (int qq = 0; qq < 2; ++qq) {
          const int within = 2 * v;
          const int i = 32 * c + within + 16 * h + qq;
          const int j = 16 * g + s;
          float val = 0.f;
          if (i < Ksz && !forb[i * Ksz + j]) val = __expf(trans[i * Ksz + j]);
          av.hh[2 * v + qq] = (_Float16)val;
        }
      A[g][c] = av.v;
    }

  // ---- exp(end) per-lane in q-layout order: ee[g*8+r] <-> j = 16g+8h+r ----
  float ee[24];
#pragma unroll
  for (int g = 0; g < 3; ++g)
#pragma unroll
    for (int r = 0; r < 8; ++r) {
      const int j = 16 * g + 8 * h + r;
      ee[g * 8 + r] = eforb[j] ? 0.f : __expf(endt[j]);
    }

  // ---- sequence length ----
  int len = 0;
  {
    const uint8_t* mrow = maskb + (size_t)bbl * Tsz;
    for (int i = 0; i < Tsz; i += 16) {
      const uint4 m4 = *(const uint4*)(mrow + i);
      len += bytesum(m4.x) + bytesum(m4.y) + bytesum(m4.z) + bytesum(m4.w);
    }
  }

  // ---- t=0: build B operand = p0^T directly in B-layout ----
  v16h Bc0, Bc1;
  int ex_acc = 0;
  float res = 0.f;  // len >= T/2 >= 2, so capture always happens at t >= 1
  {
    const float* em0 = em + (size_t)bbl * Tsz * Ksz;
    const uint8_t* tg0 = tgt + (size_t)bbl * Tsz * Ksz;
    // c0: j = 16h + i  (i=0..15);  c1: all lanes compute j = 32+i, h==1 -> 0
    float v0a[16], v1a[16];
    const uint4 tq0 = *(const uint4*)(tg0 + 16 * h);
    const uint4 tq1 = *(const uint4*)(tg0 + 32);
#pragma unroll
    for (int w = 0; w < 4; ++w) {
      const float4 e0 = *(const float4*)(em0 + 16 * h + 4 * w);
      const float4 e1 = *(const float4*)(em0 + 32 + 4 * w);
      const unsigned t0w = (w == 0) ? tq0.x : (w == 1) ? tq0.y : (w == 2) ? tq0.z : tq0.w;
      const unsigned t1w = (w == 0) ? tq1.x : (w == 1) ? tq1.y : (w == 2) ? tq1.z : tq1.w;
      const float e0f[4] = {e0.x, e0.y, e0.z, e0.w};
      const float e1f[4] = {e1.x, e1.y, e1.z, e1.w};
#pragma unroll
      for (int b = 0; b < 4; ++b) {
        const int i = 4 * w + b;
        const int j0 = 16 * h + i, j1 = 32 + i;
        const float sv0 = sforb[j0] ? -1e7f : startt[j0];
        const float sv1 = sforb[j1] ? -1e7f : startt[j1];
        const float g0 = sup ? (float)((t0w >> (8 * b)) & 0xffu) : 1.0f;
        const float g1 = sup ? (float)((t1w >> (8 * b)) & 0xffu) : 1.0f;
        v0a[i] = __expf(e0f[b] + sv0) * g0;
        v1a[i] = h0 ? __expf(e1f[b] + sv1) * g1 : 0.f;
      }
    }
    float vmax = 1e-30f;
#pragma unroll
    for (int i = 0; i < 16; ++i) vmax = fmaxf(vmax, fmaxf(v0a[i], v1a[i]));
    vmax = fmaxf(vmax, __shfl_xor(vmax, 16, 32));
    int e2;
    (void)frexpf(vmax, &e2);
    ex_acc = e2;
    union { v16h v; _Float16 hh[16]; } b0i, b1i;
#pragma unroll
    for (int i = 0; i < 16; ++i) {
      b0i.hh[i] = (_Float16)ldexpf(v0a[i], -e2);
      b1i.hh[i] = (_Float16)ldexpf(v1a[i], -e2);  // h==1 already zeroed
    }
    Bc0 = b0i.v;
    Bc1 = b1i.v;
  }

  // ---- scan with one-step-ahead prefetch, 2x unrolled double buffer ----
  const float* emL = em + (size_t)bbl * Tsz * Ksz + 8 * h;
  const uint8_t* tgL = tgt + (size_t)bbl * Tsz * Ksz + 8 * h;
  float ev0[24], ev1[24];
  uint2 gv0[3], gv1[3];
  load_step(emL, tgL, 1, ev0, gv0);
  int t = 1;
  for (; t + 1 < Tsz; t += 2) {
    load_step(emL, tgL, t + 1, ev1, gv1);
    do_step(t, ev0, gv0, A, Bc0, Bc1, ee, sup, h0, len, ex_acc, res);
    load_step(emL, tgL, (t + 2 < Tsz) ? (t + 2) : (t + 1), ev0, gv0);
    do_step(t + 1, ev1, gv1, A, Bc0, Bc1, ee, sup, h0, len, ex_acc, res);
  }
  if (t < Tsz) do_step(t, ev0, gv0, A, Bc0, Bc1, ee, sup, h0, len, ex_acc, res);

  // ---- output: -(z_sup - z_unsup) ----
  const float partner = __shfl_xor(res, 8, 32);
  if (h0 && s < 8) out[b_base + s] = partner - res;
}

extern "C" void kernel_launch(void* const* d_in, const int* in_sizes, int n_in,
                              void* d_out, int out_size, void* d_ws,
                              size_t ws_size, hipStream_t stream) {
  (void)in_sizes; (void)n_in; (void)out_size; (void)d_ws; (void)ws_size;
  const float* em = (const float*)d_in[0];
  const uint8_t* maskb = (const uint8_t*)d_in[1];
  const uint8_t* tgt = (const uint8_t*)d_in[2];
  const float* trans = (const float*)d_in[3];
  const float* startt = (const float*)d_in[4];
  const float* endt = (const float*)d_in[5];
  const uint8_t* forb = (const uint8_t*)d_in[6];
  const uint8_t* sforb = (const uint8_t*)d_in[7];
  const uint8_t* eforb = (const uint8_t*)d_in[8];
  float* out = (float*)d_out;
  // 64 tiles of (8 sequences x 2 channels) = 64 single-wave workgroups
  crf_fwd_wmma<<<dim3(Bsz / 8), dim3(32), 0, stream>>>(
      em, maskb, tgt, trans, startt, endt, forb, sforb, eforb, out);
}
